// KPConvBasicBlock_51866025066566
// MI455X (gfx1250) — matrix operand order repaired and checked by reading, not compiled, not run on hardware
//
#include <hip/hip_runtime.h>

// ---------------- problem constants (from reference) ----------------
#define N_PTS  100000
#define H      32
#define CIN    32
#define COUT   64
#define KPN    15          // kernel points (padded to 16 for WMMA M-dim)
#define WAVES  4
#define BLOCK  (WAVES * 32)
#define NTILES (N_PTS / 16)   // 6250 tiles of 16 points

typedef __attribute__((ext_vector_type(16))) __bf16 v16bf;
typedef __attribute__((ext_vector_type(8)))  float  v8f;

union BFrag {
    unsigned int u[8];
    uint4        u4[2];
    v16bf        v;
};

__device__ __forceinline__ unsigned short f2bf(float f) {
    unsigned int u = __float_as_uint(f);
    u += 0x7FFFu + ((u >> 16) & 1u);     // round-to-nearest-even
    return (unsigned short)(u >> 16);
}
__device__ __forceinline__ unsigned int pack2(float lo, float hi) {
    return (unsigned int)f2bf(lo) | ((unsigned int)f2bf(hi) << 16);
}

__global__ void __launch_bounds__(BLOCK)
kpconv_wmma_kernel(const float* __restrict__ query,
                   const float* __restrict__ support,
                   const int*   __restrict__ edge,
                   const float* __restrict__ feats,
                   const float* __restrict__ kpts,
                   const float* __restrict__ kvals,
                   float*       __restrict__ out)
{
    // K_values pre-swizzled into WMMA B-fragment layout:
    // [k][n-tile][lane][vgpr j] ; lane<16: K(c)=0..15 (pairs 2j,2j+1),
    // lane>=16: K(c)=16..31 ; N col = tile*16 + (lane&15).
    __shared__ __align__(16) unsigned int   kvfrag[KPN][4][32][8];      // 61440 B
    // per-wave weighted[point][k][c] in bf16 (A-source of 2nd GEMM)
    __shared__ __align__(16) unsigned short wtd[WAVES][16][KPN][32];    // 61440 B
    // per-wave neighbor offsets (support - query), padded float4
    __shared__ __align__(16) float4         nbr[WAVES][32];             //  2048 B

    const int tid  = threadIdx.x;
    const int lane = tid & 31;
    const int wv   = tid >> 5;
    const int half = (lane & 16) ? 1 : 0;   // upper half-wave

    // ---- stage K_values -> bf16 B-fragments in LDS (once per block) ----
    for (int e = tid; e < KPN * 4 * 32 * 8; e += BLOCK) {
        int j  = e & 7;
        int ln = (e >> 3) & 31;
        int t  = (e >> 8) & 3;
        int k  = e >> 10;
        int c0 = 2 * j + ((ln & 16) ? 16 : 0);
        int o  = t * 16 + (ln & 15);
        float f0 = kvals[(k * CIN + c0    ) * COUT + o];
        float f1 = kvals[(k * CIN + c0 + 1) * COUT + o];
        kvfrag[k][t][ln][j] = pack2(f0, f1);
    }
    __syncthreads();

    // per-lane kernel point (row of the weight matrix; fixed for the kernel)
    const int  myk    = lane & 15;
    const bool kvalid = (myk < KPN);
    float kx = 0.f, ky = 0.f, kz = 0.f;
    if (kvalid) { kx = kpts[myk * 3]; ky = kpts[myk * 3 + 1]; kz = kpts[myk * 3 + 2]; }

    // grid-stride over 16-point tiles; each wave fully owns a tile
    for (int tile = blockIdx.x * WAVES + wv; tile < NTILES;
         tile += gridDim.x * WAVES) {
        const int base = tile * 16;

        // ================= per-point GEMM1: W[16x32] x F[32x32] =========
        for (int p = 0; p < 16; ++p) {
            const int n = base + p;

            // neighbor offsets -> per-wave LDS (indices always in [0,N))
            const int idx_l = edge[n * H + lane];
            const float qx = query[n * 3 + 0];
            const float qy = query[n * 3 + 1];
            const float qz = query[n * 3 + 2];
            const float sx = support[idx_l * 3 + 0];
            const float sy = support[idx_l * 3 + 1];
            const float sz = support[idx_l * 3 + 2];
            nbr[wv][lane] = make_float4(sx - qx, sy - qy, sz - qz, 0.f);
            // wave-internal cross-lane RAW through LDS (DS ops are in-order)
            asm volatile("s_wait_dscnt 0" ::: "memory");

            // ---- A fragment: kernel-point weights, M=k row, K=h ----
            // lanes 0-15 hold K(h) = 0..15; lanes 16-31: h=16..31
            BFrag A;
            #pragma unroll
            for (int j = 0; j < 8; ++j) {
                const int h0 = 2 * j + (half ? 16 : 0);
                float4 n0 = nbr[wv][h0];
                float4 n1 = nbr[wv][h0 + 1];
                float dx = n0.x - kx, dy = n0.y - ky, dz = n0.z - kz;
                float w0 = fmaxf(1.f - sqrtf(dx * dx + dy * dy + dz * dz), 0.f);
                dx = n1.x - kx; dy = n1.y - ky; dz = n1.z - kz;
                float w1 = fmaxf(1.f - sqrtf(dx * dx + dy * dy + dz * dz), 0.f);
                if (!kvalid) { w0 = 0.f; w1 = 0.f; }   // padded row k=15
                A.u[j] = pack2(w0, w1);
            }

            // ---- B fragments: gathered features, K=h, N=c ----
            // lanes 0-15 hold K(h)=0..15 (pairs 2j,2j+1); lanes 16-31: h=16..31
            const int c = lane & 15;
            BFrag B0, B1;
            #pragma unroll
            for (int j = 0; j < 8; ++j) {
                const int h0 = 2 * j + (half ? 16 : 0);
                const int i0 = __shfl(idx_l, h0,     32);
                const int i1 = __shfl(idx_l, h0 + 1, 32);
                const float f00 = feats[i0 * CIN + c];
                const float f01 = feats[i1 * CIN + c];
                const float f10 = feats[i0 * CIN + c + 16];
                const float f11 = feats[i1 * CIN + c + 16];
                B0.u[j] = pack2(f00, f01);
                B1.u[j] = pack2(f10, f11);
            }

            v8f acc0 = {}, acc1 = {};
            acc0 = __builtin_amdgcn_wmma_f32_16x16x32_bf16(
                       false, A.v, false, B0.v, (short)0, acc0, false, false);
            acc1 = __builtin_amdgcn_wmma_f32_16x16x32_bf16(
                       false, A.v, false, B1.v, (short)0, acc1, false, false);

            // C layout: lane col = lane&15, VGPR r -> row k = r + 8*half
            #pragma unroll
            for (int r = 0; r < 8; ++r) {
                const int kk = r + (half ? 8 : 0);
                if (kk < KPN) {
                    wtd[wv][p][kk][c]      = f2bf(acc0[r]);
                    wtd[wv][p][kk][c + 16] = f2bf(acc1[r]);
                }
            }
        }

        asm volatile("s_wait_dscnt 0" ::: "memory");

        // ======= GEMM2: out[16x64] = weighted[16x(15*32)] x Kv =========
        v8f acc[4] = {{}, {}, {}, {}};
        const int pt = lane & 15;               // A-matrix row = point
        for (int k = 0; k < KPN; ++k) {
            BFrag A2;                           // K-dim = channel c of chunk k
            #pragma unroll
            for (int j = 0; j < 8; ++j) {
                const int c0 = 2 * j + (half ? 16 : 0);
                A2.u[j] = *(const unsigned int*)&wtd[wv][pt][k][c0];
            }
            #pragma unroll
            for (int t = 0; t < 4; ++t) {
                BFrag B2;
                const uint4* bp = (const uint4*)&kvfrag[k][t][lane][0];
                B2.u4[0] = bp[0];
                B2.u4[1] = bp[1];
                acc[t] = __builtin_amdgcn_wmma_f32_16x16x32_bf16(
                             false, A2.v, false, B2.v, (short)0, acc[t],
                             false, false);
            }
        }

        // ---- write out: lane col = t*16+(lane&15), VGPR r -> point row ----
        #pragma unroll
        for (int t = 0; t < 4; ++t) {
            const int o = t * 16 + (lane & 15);
            #pragma unroll
            for (int r = 0; r < 8; ++r) {
                const int prow = r + (half ? 8 : 0);
                out[(base + prow) * COUT + o] = acc[t][r];
            }
        }
    }
}

extern "C" void kernel_launch(void* const* d_in, const int* in_sizes, int n_in,
                              void* d_out, int out_size, void* d_ws, size_t ws_size,
                              hipStream_t stream) {
    const float* query   = (const float*)d_in[0];
    const float* support = (const float*)d_in[1];
    const int*   edge    = (const int*)  d_in[2];
    const float* feats   = (const float*)d_in[3];
    const float* kpts    = (const float*)d_in[4];
    const float* kvals   = (const float*)d_in[5];
    float*       out     = (float*)d_out;

    int maxBlocks = (NTILES + WAVES - 1) / WAVES;
    int blocks = maxBlocks < 512 ? maxBlocks : 512;
    kpconv_wmma_kernel<<<blocks, BLOCK, 0, stream>>>(
        query, support, edge, feats, kpts, kvals, out);
}
